// MonitoredCausalSelfAttention_69535520522343
// MI455X (gfx1250) — matrix operand order repaired and checked by reading, not compiled, not run on hardware
//
#include <hip/hip_runtime.h>

// ---------------------------------------------------------------------------
// MonitoredCausalSelfAttention for MI455X (gfx1250, wave32, WMMA bf16)
// B=2, T=2048, C=1024, H=16, D=64
// ---------------------------------------------------------------------------

typedef __bf16 bf16_t;
typedef __attribute__((ext_vector_type(16))) __bf16 v16bf;
typedef __attribute__((ext_vector_type(8)))  __bf16 v8bf;
typedef __attribute__((ext_vector_type(8)))  float  v8f;

union FragBF { v16bf v; v8bf h[2]; };

#define WMMA_BF16(A, B, C) \
  __builtin_amdgcn_wmma_f32_16x16x32_bf16(false, (A), false, (B), (short)0, (C), false, false)

static __device__ __forceinline__ bf16_t tobf(float f) {
  union { float f; unsigned u; } in; in.f = f;
  unsigned u = in.u;
  u += 0x7FFFu + ((u >> 16) & 1u);        // round-to-nearest-even
  union { unsigned short s; bf16_t b; } out;
  out.s = (unsigned short)(u >> 16);
  return out.b;
}

static __device__ __forceinline__ v8f vzero() {
  v8f z = {0.f, 0.f, 0.f, 0.f, 0.f, 0.f, 0.f, 0.f};
  return z;
}

// ---------------------------------------------------------------------------
// Kernel 0a: fp32 -> bf16 elementwise
// ---------------------------------------------------------------------------
__global__ void cvt_f32_bf16(const float* __restrict__ src, bf16_t* __restrict__ dst, int n) {
  int stride = gridDim.x * blockDim.x;
  for (int i = blockIdx.x * blockDim.x + threadIdx.x; i < n; i += stride)
    dst[i] = tobf(src[i]);
}

// Kernel 0b: fp32 [K][N] -> bf16 [N][K] (transpose + convert)
__global__ void tcvt_f32_bf16(const float* __restrict__ src, bf16_t* __restrict__ dst,
                              int K, int N) {
  int total = K * N;
  int stride = gridDim.x * blockDim.x;
  for (int i = blockIdx.x * blockDim.x + threadIdx.x; i < total; i += stride) {
    int k = i / N;
    int n = i - k * N;
    dst[(size_t)n * K + k] = tobf(src[i]);
  }
}

// ---------------------------------------------------------------------------
// Kernel 1: QKV GEMM  [4096,1024] x [1024,3072] + bias
//   Block tile 128x128, BK=32, 8 waves (2x4), each wave 64x32 = 4x2 WMMA tiles.
//   Epilogue routes q,k -> bf16 [B,H,T,D]; v -> bf16 [B,H,D,T]; k,v fp32 -> kv out.
// ---------------------------------------------------------------------------
__global__ __launch_bounds__(256) void qkv_gemm(
    const bf16_t* __restrict__ xbf,     // [4096][1024]
    const bf16_t* __restrict__ waT,     // [3072][1024]  (w_attn transposed)
    const float*  __restrict__ battn,   // [3072]
    bf16_t* __restrict__ qbf,           // [B,H,T,D]
    bf16_t* __restrict__ kbf,           // [B,H,T,D]
    bf16_t* __restrict__ vtbf,          // [B,H,D,T]
    float*  __restrict__ kvout)         // [2,B,H,T,D] fp32 (output region)
{
  const int K = 1024;
  const int m0 = (blockIdx.x / 24) * 128;
  const int n0 = (blockIdx.x % 24) * 128;

  __shared__ __align__(16) bf16_t As[128 * 40];  // 32 cols + 8 pad
  __shared__ __align__(16) bf16_t Bs[128 * 40];

  const int tid  = threadIdx.x;
  const int wave = tid >> 5, lane = tid & 31;
  const int lo = lane & 15, hi = lane >> 4;
  const int wm = wave >> 2, wn = wave & 3;

  v8f acc[4][2];
#pragma unroll
  for (int mt = 0; mt < 4; ++mt)
#pragma unroll
    for (int nt = 0; nt < 2; ++nt) acc[mt][nt] = vzero();

  const int ldrow = tid >> 1, ldhalf = tid & 1;

  for (int kb = 0; kb < K; kb += 32) {
    // global -> regs
    const bf16_t* sa = xbf + (size_t)(m0 + ldrow) * K + kb + ldhalf * 16;
    const bf16_t* sb = waT + (size_t)(n0 + ldrow) * K + kb + ldhalf * 16;
    v8bf a0 = *(const v8bf*)sa;
    v8bf a1 = *(const v8bf*)(sa + 8);
    v8bf b0 = *(const v8bf*)sb;
    v8bf b1 = *(const v8bf*)(sb + 8);

    __syncthreads();
    *(v8bf*)(As + ldrow * 40 + ldhalf * 16)     = a0;
    *(v8bf*)(As + ldrow * 40 + ldhalf * 16 + 8) = a1;
    *(v8bf*)(Bs + ldrow * 40 + ldhalf * 16)     = b0;
    *(v8bf*)(Bs + ldrow * 40 + ldhalf * 16 + 8) = b1;
    __syncthreads();

    FragBF af[4], bfr[2];
#pragma unroll
    for (int mt = 0; mt < 4; ++mt) {
      int row = wm * 64 + mt * 16 + lo;
      af[mt].h[0] = *(const v8bf*)(As + row * 40 + 8 * hi);
      af[mt].h[1] = *(const v8bf*)(As + row * 40 + 16 + 8 * hi);
    }
#pragma unroll
    for (int nt = 0; nt < 2; ++nt) {
      int n = wn * 32 + nt * 16 + lo;
      bfr[nt].h[0] = *(const v8bf*)(Bs + n * 40 + 8 * hi);
      bfr[nt].h[1] = *(const v8bf*)(Bs + n * 40 + 16 + 8 * hi);
    }
#pragma unroll
    for (int mt = 0; mt < 4; ++mt)
#pragma unroll
      for (int nt = 0; nt < 2; ++nt)
        acc[mt][nt] = WMMA_BF16(af[mt].v, bfr[nt].v, acc[mt][nt]);
  }

  // Epilogue: C layout = lane(lo)->col, VGPR r -> row 8*hi+r
#pragma unroll
  for (int nt = 0; nt < 2; ++nt) {
    const int n = n0 + wn * 32 + nt * 16 + lo;
    const float bv = battn[n];
#pragma unroll
    for (int mt = 0; mt < 4; ++mt) {
#pragma unroll
      for (int r = 0; r < 8; ++r) {
        int m = m0 + wm * 64 + mt * 16 + 8 * hi + r;
        float val = acc[mt][nt][r] + bv;
        int b = m >> 11, t = m & 2047;
        if (n < 1024) {
          int h = n >> 6, d = n & 63;
          qbf[(((size_t)(b * 16 + h)) * 2048 + t) * 64 + d] = tobf(val);
        } else if (n < 2048) {
          int n2 = n - 1024, h = n2 >> 6, d = n2 & 63;
          size_t idx = (((size_t)(b * 16 + h)) * 2048 + t) * 64 + d;
          kbf[idx] = tobf(val);
          kvout[idx] = val;                       // kv[0] = k
        } else {
          int n3 = n - 2048, h = n3 >> 6, d = n3 & 63;
          vtbf[(((size_t)(b * 16 + h)) * 64 + d) * 2048 + t] = tobf(val);
          kvout[(((size_t)(b * 16 + h)) * 2048 + t) * 64 + d + 4194304] = val;  // kv[1] = v
        }
      }
    }
  }
}

// ---------------------------------------------------------------------------
// Kernel 2: flash attention + entropy.
//   Grid: B*H*(T/64) blocks of 128 threads (4 independent waves, 16 q-rows each).
//   Online softmax tracks m, Z, S1 = sum p*(s-m); entropy = log Z - S1/Z.
// ---------------------------------------------------------------------------
__global__ __launch_bounds__(128) void attn_kernel(
    const bf16_t* __restrict__ qbf,   // [B,H,T,D]
    const bf16_t* __restrict__ kbf,   // [B,H,T,D]
    const bf16_t* __restrict__ vtbf,  // [B,H,D,T]
    bf16_t* __restrict__ ybf,         // [B,T,C]
    float*  __restrict__ partials)    // [B*H*(T/16)]
{
  const int T = 2048;
  const int bh   = blockIdx.x >> 5;       // /32 q-blocks
  const int qblk = blockIdx.x & 31;
  const int wave = threadIdx.x >> 5, lane = threadIdx.x & 31;
  const int lo = lane & 15, hi = lane >> 4;
  const int q0 = qblk * 64 + wave * 16;
  const int b = bh >> 4, h = bh & 15;

  const bf16_t* qb  = qbf  + (size_t)bh * T * 64;
  const bf16_t* kb  = kbf  + (size_t)bh * T * 64;
  const bf16_t* vtb = vtbf + (size_t)bh * 64 * T;

  __shared__ __align__(16) bf16_t plds[4][16 * 32];
  bf16_t* pw = &plds[wave][0];

  // Preload Q fragments (A layout): row=lo, chunks at k = 8*hi and 16+8*hi
  FragBF qf[2];
#pragma unroll
  for (int c = 0; c < 2; ++c) {
    const bf16_t* qp = qb + (size_t)(q0 + lo) * 64 + c * 32 + 8 * hi;
    qf[c].h[0] = *(const v8bf*)qp;
    qf[c].h[1] = *(const v8bf*)(qp + 16);
  }

  v8f O[4];
#pragma unroll
  for (int nt = 0; nt < 4; ++nt) O[nt] = vzero();
  float mrow[8], zrow[8], s1row[8];
#pragma unroll
  for (int r = 0; r < 8; ++r) { mrow[r] = -1e30f; zrow[r] = 0.f; s1row[r] = 0.f; }

  const int nchunk = (q0 + 16 + 31) >> 5;   // keys 0 .. q0+15

  for (int j = 0; j < nchunk; ++j) {
    const int kbase = j * 32;

    // S = Q @ K^T (two 16-col tiles; B layout: col=lo=key, chunks at k = 8*hi and 16+8*hi)
    v8f S[2];
#pragma unroll
    for (int nt = 0; nt < 2; ++nt) {
      v8f s = vzero();
#pragma unroll
      for (int c = 0; c < 2; ++c) {
        FragBF kf;
        const bf16_t* kp = kb + (size_t)(kbase + nt * 16 + lo) * 64 + c * 32 + 8 * hi;
        kf.h[0] = *(const v8bf*)kp;
        kf.h[1] = *(const v8bf*)(kp + 16);
        s = WMMA_BF16(qf[c].v, kf.v, s);
      }
      S[nt] = s;
    }

    // scale + causal mask
#pragma unroll
    for (int nt = 0; nt < 2; ++nt) {
#pragma unroll
      for (int r = 0; r < 8; ++r) {
        float v = S[nt][r] * 0.125f;     // 1/sqrt(64)
        int key = kbase + nt * 16 + lo;
        int qi  = q0 + 8 * hi + r;
        if (key > qi) v = -1e30f;
        S[nt][r] = v;
      }
    }

    // online softmax + entropy stats (row = 8*hi + r; 16 lanes share a row)
#pragma unroll
    for (int r = 0; r < 8; ++r) {
      float rm = fmaxf(S[0][r], S[1][r]);
      rm = fmaxf(rm, __shfl_xor(rm, 1, 32));
      rm = fmaxf(rm, __shfl_xor(rm, 2, 32));
      rm = fmaxf(rm, __shfl_xor(rm, 4, 32));
      rm = fmaxf(rm, __shfl_xor(rm, 8, 32));
      float nm = fmaxf(mrow[r], rm);
      float d0 = S[0][r] - nm, d1 = S[1][r] - nm;
      float p0 = __expf(d0), p1 = __expf(d1);
      float sum = p0 + p1;
      float dot = p0 * d0 + p1 * d1;
      sum += __shfl_xor(sum, 1, 32); dot += __shfl_xor(dot, 1, 32);
      sum += __shfl_xor(sum, 2, 32); dot += __shfl_xor(dot, 2, 32);
      sum += __shfl_xor(sum, 4, 32); dot += __shfl_xor(dot, 4, 32);
      sum += __shfl_xor(sum, 8, 32); dot += __shfl_xor(dot, 8, 32);
      float al = __expf(mrow[r] - nm);
      s1row[r] = al * (s1row[r] + (mrow[r] - nm) * zrow[r]) + dot;
      zrow[r]  = zrow[r] * al + sum;
      mrow[r]  = nm;
#pragma unroll
      for (int nt = 0; nt < 4; ++nt) O[nt][r] *= al;
      int row = 8 * hi + r;
      pw[row * 32 + lo]      = tobf(p0);
      pw[row * 32 + 16 + lo] = tobf(p1);
    }

    // P (C layout in LDS) -> A fragment; same-wave DS ops are in-order
    FragBF pf;
    pf.h[0] = *(const v8bf*)(pw + lo * 32 + 8 * hi);
    pf.h[1] = *(const v8bf*)(pw + lo * 32 + 16 + 8 * hi);

    // O += P @ V  (B layout from transposed V: col=lo=d, key chunks at k = 8*hi and 16+8*hi)
#pragma unroll
    for (int nt = 0; nt < 4; ++nt) {
      FragBF vf;
      const bf16_t* vp = vtb + (size_t)(nt * 16 + lo) * T + kbase + 8 * hi;
      vf.h[0] = *(const v8bf*)vp;
      vf.h[1] = *(const v8bf*)(vp + 16);
      O[nt] = WMMA_BF16(pf.v, vf.v, O[nt]);
    }
  }

  // normalize + write y (bf16, [B,T,C] with c = h*64 + d)
#pragma unroll
  for (int r = 0; r < 8; ++r) {
    float inv = 1.f / zrow[r];
    int q = q0 + 8 * hi + r;
#pragma unroll
    for (int nt = 0; nt < 4; ++nt) {
      float val = O[nt][r] * inv;
      ybf[((size_t)(b * 2048 + q)) * 1024 + h * 64 + nt * 16 + lo] = tobf(val);
    }
  }

  // entropy partial: H_row = log Z - S1/Z
  float e = 0.f;
  if (lo == 0) {
#pragma unroll
    for (int r = 0; r < 8; ++r) e += __logf(zrow[r]) - s1row[r] / zrow[r];
  }
  e += __shfl_xor(e, 16, 32);
  if (lane == 0) partials[blockIdx.x * 4 + wave] = e;
}

// ---------------------------------------------------------------------------
// Kernel 3: projection GEMM  y[4096,1024] = ybf @ w_projT + b_proj  (fp32 out)
// ---------------------------------------------------------------------------
__global__ __launch_bounds__(256) void proj_gemm(
    const bf16_t* __restrict__ ybf,     // [4096][1024]
    const bf16_t* __restrict__ wpT,     // [1024][1024]
    const float*  __restrict__ bproj,   // [1024]
    float* __restrict__ out)            // [4096][1024] fp32
{
  const int K = 1024;
  const int m0 = (blockIdx.x >> 3) * 128;
  const int n0 = (blockIdx.x & 7) * 128;

  __shared__ __align__(16) bf16_t As[128 * 40];
  __shared__ __align__(16) bf16_t Bs[128 * 40];

  const int tid  = threadIdx.x;
  const int wave = tid >> 5, lane = tid & 31;
  const int lo = lane & 15, hi = lane >> 4;
  const int wm = wave >> 2, wn = wave & 3;

  v8f acc[4][2];
#pragma unroll
  for (int mt = 0; mt < 4; ++mt)
#pragma unroll
    for (int nt = 0; nt < 2; ++nt) acc[mt][nt] = vzero();

  const int ldrow = tid >> 1, ldhalf = tid & 1;

  for (int kb = 0; kb < K; kb += 32) {
    const bf16_t* sa = ybf + (size_t)(m0 + ldrow) * K + kb + ldhalf * 16;
    const bf16_t* sb = wpT + (size_t)(n0 + ldrow) * K + kb + ldhalf * 16;
    v8bf a0 = *(const v8bf*)sa;
    v8bf a1 = *(const v8bf*)(sa + 8);
    v8bf b0 = *(const v8bf*)sb;
    v8bf b1 = *(const v8bf*)(sb + 8);

    __syncthreads();
    *(v8bf*)(As + ldrow * 40 + ldhalf * 16)     = a0;
    *(v8bf*)(As + ldrow * 40 + ldhalf * 16 + 8) = a1;
    *(v8bf*)(Bs + ldrow * 40 + ldhalf * 16)     = b0;
    *(v8bf*)(Bs + ldrow * 40 + ldhalf * 16 + 8) = b1;
    __syncthreads();

    FragBF af[4], bfr[2];
#pragma unroll
    for (int mt = 0; mt < 4; ++mt) {
      int row = wm * 64 + mt * 16 + lo;
      af[mt].h[0] = *(const v8bf*)(As + row * 40 + 8 * hi);
      af[mt].h[1] = *(const v8bf*)(As + row * 40 + 16 + 8 * hi);
    }
#pragma unroll
    for (int nt = 0; nt < 2; ++nt) {
      int n = wn * 32 + nt * 16 + lo;
      bfr[nt].h[0] = *(const v8bf*)(Bs + n * 40 + 8 * hi);
      bfr[nt].h[1] = *(const v8bf*)(Bs + n * 40 + 16 + 8 * hi);
    }
#pragma unroll
    for (int mt = 0; mt < 4; ++mt)
#pragma unroll
      for (int nt = 0; nt < 2; ++nt)
        acc[mt][nt] = WMMA_BF16(af[mt].v, bfr[nt].v, acc[mt][nt]);
  }

#pragma unroll
  for (int nt = 0; nt < 2; ++nt) {
    const int n = n0 + wn * 32 + nt * 16 + lo;
    const float bv = bproj[n];
#pragma unroll
    for (int mt = 0; mt < 4; ++mt) {
#pragma unroll
      for (int r = 0; r < 8; ++r) {
        int m = m0 + wm * 64 + mt * 16 + 8 * hi + r;
        out[(size_t)m * 1024 + n] = acc[mt][nt][r] + bv;
      }
    }
  }
}

// ---------------------------------------------------------------------------
// Kernel 4: entropy reduction (deterministic)
// ---------------------------------------------------------------------------
__global__ void ent_reduce(const float* __restrict__ partials, int n,
                           float* __restrict__ out_ent) {
  __shared__ float sm[256];
  float s = 0.f;
  for (int i = threadIdx.x; i < n; i += 256) s += partials[i];
  sm[threadIdx.x] = s;
  __syncthreads();
  for (int st = 128; st > 0; st >>= 1) {
    if (threadIdx.x < st) sm[threadIdx.x] += sm[threadIdx.x + st];
    __syncthreads();
  }
  if (threadIdx.x == 0) *out_ent = sm[0] * (1.0f / 65536.0f);  // / (B*H*T)
}

// ---------------------------------------------------------------------------
// Host launcher
// ---------------------------------------------------------------------------
extern "C" void kernel_launch(void* const* d_in, const int* in_sizes, int n_in,
                              void* d_out, int out_size, void* d_ws, size_t ws_size,
                              hipStream_t stream) {
  const float* x      = (const float*)d_in[0];   // [2,2048,1024]
  const float* w_attn = (const float*)d_in[1];   // [1024,3072]
  const float* b_attn = (const float*)d_in[2];   // [3072]
  const float* w_proj = (const float*)d_in[3];   // [1024,1024]
  const float* b_proj = (const float*)d_in[4];   // [1024]
  float* out = (float*)d_out;                    // y(4194304) | ent(1) | kv(8388608)

  char* ws = (char*)d_ws;
  bf16_t* xbf  = (bf16_t*)(ws);
  bf16_t* waT  = (bf16_t*)(ws + 8388608);
  bf16_t* wpT  = (bf16_t*)(ws + 14680064);
  bf16_t* qbf  = (bf16_t*)(ws + 16777216);
  bf16_t* kbf  = (bf16_t*)(ws + 25165824);
  bf16_t* vtbf = (bf16_t*)(ws + 33554432);
  bf16_t* ybf  = (bf16_t*)(ws + 41943040);
  float*  part = (float*)(ws + 50331648);

  float* y_out  = out;
  float* ent    = out + 4194304;
  float* kv_out = out + 4194305;

  cvt_f32_bf16 <<<2048, 256, 0, stream>>>(x, xbf, 4194304);
  tcvt_f32_bf16<<<3072, 256, 0, stream>>>(w_attn, waT, 1024, 3072);
  tcvt_f32_bf16<<<1024, 256, 0, stream>>>(w_proj, wpT, 1024, 1024);

  qkv_gemm<<<768, 256, 0, stream>>>(xbf, waT, b_attn, qbf, kbf, vtbf, kv_out);

  attn_kernel<<<1024, 128, 0, stream>>>(qbf, kbf, vtbf, ybf, part);

  proj_gemm<<<256, 256, 0, stream>>>(ybf, wpT, b_proj, y_out);

  ent_reduce<<<1, 256, 0, stream>>>(part, 4096, ent);
}
